// CausalSelfAttention_5763846111798
// MI455X (gfx1250) — compile-verified, matrix operand order checked by reading, not run on hardware
//
#include <hip/hip_runtime.h>

typedef __bf16 bf16_t;
typedef __attribute__((ext_vector_type(16))) __bf16        v16bf;
typedef __attribute__((ext_vector_type(8)))  float         v8f;
typedef __attribute__((ext_vector_type(4)))  unsigned int  u32x4;

#define B_    16
#define T_    512
#define C_    1080
#define H_    20
#define HD_   54
#define HDP_  64
#define KP_   1088          // C padded to multiple of 32
#define N1_   3240          // 3*C
#define M_    (B_*T_)       // 8192

// ---------------------------------------------------------------- WMMA helpers

__device__ __forceinline__ v8f wmma_bf16(v16bf a, v16bf b, v8f c) {
  // D = A(16x32 bf16) * B(32x16 bf16) + C(16x16 f32)
  return __builtin_amdgcn_wmma_f32_16x16x32_bf16(false, a, false, b,
                                                 (short)0, c, false, false);
}

// Per-lane bf16 16x32 fragment: two contiguous 16-byte chunks
// (K = half*8..half*8+7 and K = 16+half*8..16+half*8+7).
__device__ __forceinline__ v16bf frag16(const bf16_t* p0, const bf16_t* p1) {
  union U { v16bf v; u32x4 q[2]; } f;
  f.q[0] = *reinterpret_cast<const u32x4*>(p0);
  f.q[1] = *reinterpret_cast<const u32x4*>(p1);
  return f.v;
}

__device__ __forceinline__ float rowmax16(float x) {
  #pragma unroll
  for (int m = 1; m < 16; m <<= 1) x = fmaxf(x, __shfl_xor(x, m, 32));
  return x;
}
__device__ __forceinline__ float rowsum16(float x) {
  #pragma unroll
  for (int m = 1; m < 16; m <<= 1) x += __shfl_xor(x, m, 32);
  return x;
}

// ---------------------------------------------------------------- prep kernels

__global__ __launch_bounds__(256) void zero_u128_k(u32x4* __restrict__ p, int n) {
  int i = blockIdx.x * 256 + threadIdx.x;
  if (i < n) { u32x4 z = {0u, 0u, 0u, 0u}; p[i] = z; }
}

// x [M,C] f32 -> xb [M,KP] bf16, zero pad K
__global__ __launch_bounds__(256) void conv_x_k(const float* __restrict__ x,
                                                bf16_t* __restrict__ xb) {
  int idx = blockIdx.x * 256 + threadIdx.x;
  if (idx >= M_ * KP_) return;
  int m = idx / KP_, k = idx - m * KP_;
  xb[idx] = (k < C_) ? (bf16_t)x[(size_t)m * C_ + k] : (bf16_t)0.0f;
}

// W [C,ncols] f32 -> Wt [ncols,KP] bf16 (transposed, zero pad K)
__global__ __launch_bounds__(256) void conv_wT_k(const float* __restrict__ w,
                                                 bf16_t* __restrict__ wt,
                                                 int ncols, int total) {
  int idx = blockIdx.x * 256 + threadIdx.x;
  if (idx >= total) return;
  int n = idx / KP_, k = idx - n * KP_;
  wt[idx] = (k < C_) ? (bf16_t)w[(size_t)k * ncols + n] : (bf16_t)0.0f;
}

// ---------------------------------------------------------------- GEMM 1: QKV
// [M,KP] x [KP,N1] -> scatter q/k/v (bf16, head-padded; v transposed [d,t]).
// Block: 256 thr = 8 waves (2 along M x 4 along N); block tile 64x256.
// Wave tile 32x64 = 2x4 WMMA subtiles: 6 b128-pair loads per K-step, 8 WMMAs.
// launch_bounds(256,1) relaxes the VGPR budget so the scheduler can keep
// multiple load octets in flight across unrolled K-steps.
__global__ __launch_bounds__(256, 1)
void gemm_qkv_k(const bf16_t* __restrict__ xb, const bf16_t* __restrict__ waT,
                const float* __restrict__ b_attn,
                bf16_t* __restrict__ qb, bf16_t* __restrict__ kb,
                bf16_t* __restrict__ vb) {
  const int lane = threadIdx.x & 31, wave = threadIdx.x >> 5;
  const int wm = wave & 1, wn = wave >> 1;
  const int bm = (int)blockIdx.x * 64 + wm * 32;
  const int bn = (int)blockIdx.y * 256 + wn * 64;
  const int r = lane & 15, half = lane >> 4;

  v8f acc[2][4];
  #pragma unroll
  for (int a = 0; a < 2; ++a)
    #pragma unroll
    for (int bq = 0; bq < 4; ++bq) acc[a][bq] = (v8f)(0.0f);

  const bf16_t* arow0 = xb + (size_t)(bm + r) * KP_;
  const bf16_t* arow1 = xb + (size_t)(bm + 16 + r) * KP_;
  const bf16_t* brow[4];
  #pragma unroll
  for (int j = 0; j < 4; ++j) {
    int n = bn + j * 16 + r;
    if (n > N1_ - 1) n = N1_ - 1;
    brow[j] = waT + (size_t)n * KP_;
  }

  for (int kk = 0; kk < KP_; kk += 32) {
    // Speculative prefetch a few K-steps ahead (dropped silently if OOB).
    __builtin_prefetch(arow0 + kk + 256, 0, 1);
    __builtin_prefetch(arow1 + kk + 256, 0, 1);

    const int o0 = kk + half * 8, o1 = kk + 16 + half * 8;
    v16bf a0 = frag16(arow0 + o0, arow0 + o1);
    v16bf a1 = frag16(arow1 + o0, arow1 + o1);
    #pragma unroll
    for (int j = 0; j < 4; ++j) {
      v16bf bj = frag16(brow[j] + o0, brow[j] + o1);
      acc[0][j] = wmma_bf16(a0, bj, acc[0][j]);
      acc[1][j] = wmma_bf16(a1, bj, acc[1][j]);
    }
  }

  // Epilogue: bias + scatter into per-head padded layouts.
  #pragma unroll
  for (int ms = 0; ms < 2; ++ms) {
    #pragma unroll
    for (int ns = 0; ns < 4; ++ns) {
      int ncol = bn + ns * 16 + r;
      if (ncol >= N1_) continue;
      int which = ncol / C_;
      int rr = ncol - which * C_;
      int h = rr / HD_, d = rr - h * HD_;
      float bias = b_attn[ncol];
      #pragma unroll
      for (int i = 0; i < 8; ++i) {
        int m = bm + ms * 16 + i + 8 * half;
        int b = m >> 9, t = m & (T_ - 1);
        float val = acc[ms][ns][i] + bias;
        size_t bh = (size_t)(b * H_ + h);
        if (which == 0)      qb[(bh * T_ + t) * HDP_ + d]  = (bf16_t)val;
        else if (which == 1) kb[(bh * T_ + t) * HDP_ + d]  = (bf16_t)val;
        else                 vb[(bh * HDP_ + d) * T_ + t]  = (bf16_t)val; // transposed
      }
    }
  }
}

// ---------------------------------------------------------------- attention
// One wave per 16-row q tile; flash-style online softmax over 32-key tiles.
__global__ __launch_bounds__(256)
void attn_k(const bf16_t* __restrict__ qb, const bf16_t* __restrict__ kb,
            const bf16_t* __restrict__ vb, bf16_t* __restrict__ yb) {
  __shared__ bf16_t lds_p[8][16 * 32];   // per-wave P staging (bf16 16x32)

  const int lane = threadIdx.x & 31, wave = threadIdx.x >> 5;
  const int tile = (int)blockIdx.x * 8 + wave;       // B*H*(T/16) tiles
  const int bh = tile >> 5;                           // T/16 == 32
  const int qt = tile & 31, q0 = qt * 16;
  const int b = bh / H_, h = bh - b * H_;
  const int r = lane & 15, half = lane >> 4;

  const bf16_t* qbase = qb + (size_t)bh * T_ * HDP_;
  const bf16_t* kbase = kb + (size_t)bh * T_ * HDP_;
  const bf16_t* vbase = vb + (size_t)bh * HDP_ * T_;

  const bf16_t* qrow = qbase + (size_t)(q0 + r) * HDP_;
  v16bf aq0 = frag16(qrow + half * 8,      qrow + 16 + half * 8);
  v16bf aq1 = frag16(qrow + 32 + half * 8, qrow + 48 + half * 8);

  v8f o[4];
  #pragma unroll
  for (int j = 0; j < 4; ++j) o[j] = (v8f)(0.0f);
  float mrow[8], lrow[8];
  #pragma unroll
  for (int i = 0; i < 8; ++i) { mrow[i] = -1e30f; lrow[i] = 0.0f; }

  const float scale = 0.13608276348795434f;  // 1/sqrt(54)
  bf16_t* pl = lds_p[wave];
  const int ntiles = (q0 + 16 + 31) >> 5;

  for (int kt = 0; kt < ntiles; ++kt) {
    const int k0 = kt * 32;
    v8f c0 = (v8f)(0.0f), c1 = (v8f)(0.0f);

    const bf16_t* kr0 = kbase + (size_t)(k0 + r) * HDP_;
    const bf16_t* kr1 = kbase + (size_t)(k0 + 16 + r) * HDP_;
    v16bf bk00 = frag16(kr0 + half * 8,      kr0 + 16 + half * 8);
    v16bf bk01 = frag16(kr0 + 32 + half * 8, kr0 + 48 + half * 8);
    v16bf bk10 = frag16(kr1 + half * 8,      kr1 + 16 + half * 8);
    v16bf bk11 = frag16(kr1 + 32 + half * 8, kr1 + 48 + half * 8);
    c0 = wmma_bf16(aq0, bk00, c0);  c0 = wmma_bf16(aq1, bk01, c0);
    c1 = wmma_bf16(aq0, bk10, c1);  c1 = wmma_bf16(aq1, bk11, c1);

    const int col0 = k0 + r, col1 = k0 + 16 + r;
    #pragma unroll
    for (int i = 0; i < 8; ++i) {
      const int qr = q0 + i + 8 * half;
      float s0 = (col0 <= qr) ? c0[i] * scale : -1e30f;
      float s1 = (col1 <= qr) ? c1[i] * scale : -1e30f;
      float mx = rowmax16(fmaxf(s0, s1));
      float mnew = fmaxf(mrow[i], mx);
      float sc = __expf(mrow[i] - mnew);
      float p0 = __expf(s0 - mnew);
      float p1 = __expf(s1 - mnew);
      lrow[i] = lrow[i] * sc + rowsum16(p0 + p1);
      mrow[i] = mnew;
      o[0][i] *= sc; o[1][i] *= sc; o[2][i] *= sc; o[3][i] *= sc;
      const int rowl = i + 8 * half;
      pl[rowl * 32 + r]      = (bf16_t)p0;
      pl[rowl * 32 + 16 + r] = (bf16_t)p1;
    }
    asm volatile("s_wait_dscnt 0" ::: "memory");   // P stores visible wave-wide

    const bf16_t* prow = pl + r * 32;
    v16bf ap = frag16(prow + half * 8, prow + 16 + half * 8);
    #pragma unroll
    for (int j = 0; j < 4; ++j) {
      const bf16_t* vr = vbase + (size_t)(j * 16 + r) * T_ + k0;
      v16bf bv = frag16(vr + half * 8, vr + 16 + half * 8);
      o[j] = wmma_bf16(ap, bv, o[j]);
    }
  }

  // Normalize + store into padded y (cols h*54+d, d<54).
  #pragma unroll
  for (int j = 0; j < 4; ++j) {
    const int d = j * 16 + r;
    if (d >= HD_) continue;
    const int col = h * HD_ + d;
    #pragma unroll
    for (int i = 0; i < 8; ++i) {
      const int m = b * T_ + q0 + i + 8 * half;
      yb[(size_t)m * KP_ + col] = (bf16_t)(o[j][i] / lrow[i]);
    }
  }
}

// ---------------------------------------------------------------- GEMM 2: proj
// Same 64x256 block / 32x64 wave tiling as GEMM 1; bias + f32 store epilogue.
__global__ __launch_bounds__(256, 1)
void gemm_proj_k(const bf16_t* __restrict__ yb, const bf16_t* __restrict__ wpT,
                 const float* __restrict__ b_proj, float* __restrict__ out) {
  const int lane = threadIdx.x & 31, wave = threadIdx.x >> 5;
  const int wm = wave & 1, wn = wave >> 1;
  const int bm = (int)blockIdx.x * 64 + wm * 32;
  const int bn = (int)blockIdx.y * 256 + wn * 64;
  const int r = lane & 15, half = lane >> 4;

  v8f acc[2][4];
  #pragma unroll
  for (int a = 0; a < 2; ++a)
    #pragma unroll
    for (int bq = 0; bq < 4; ++bq) acc[a][bq] = (v8f)(0.0f);

  const bf16_t* arow0 = yb + (size_t)(bm + r) * KP_;
  const bf16_t* arow1 = yb + (size_t)(bm + 16 + r) * KP_;
  const bf16_t* brow[4];
  #pragma unroll
  for (int j = 0; j < 4; ++j) {
    int n = bn + j * 16 + r;
    if (n > C_ - 1) n = C_ - 1;
    brow[j] = wpT + (size_t)n * KP_;
  }

  for (int kk = 0; kk < KP_; kk += 32) {
    __builtin_prefetch(arow0 + kk + 256, 0, 1);
    __builtin_prefetch(arow1 + kk + 256, 0, 1);

    const int o0 = kk + half * 8, o1 = kk + 16 + half * 8;
    v16bf a0 = frag16(arow0 + o0, arow0 + o1);
    v16bf a1 = frag16(arow1 + o0, arow1 + o1);
    #pragma unroll
    for (int j = 0; j < 4; ++j) {
      v16bf bj = frag16(brow[j] + o0, brow[j] + o1);
      acc[0][j] = wmma_bf16(a0, bj, acc[0][j]);
      acc[1][j] = wmma_bf16(a1, bj, acc[1][j]);
    }
  }

  #pragma unroll
  for (int ms = 0; ms < 2; ++ms) {
    #pragma unroll
    for (int ns = 0; ns < 4; ++ns) {
      int ncol = bn + ns * 16 + r;
      if (ncol >= C_) continue;
      float bias = b_proj[ncol];
      #pragma unroll
      for (int i = 0; i < 8; ++i) {
        int m = bm + ms * 16 + i + 8 * half;
        out[(size_t)m * C_ + ncol] = acc[ms][ns][i] + bias;
      }
    }
  }
}

// ---------------------------------------------------------------- host launch

extern "C" void kernel_launch(void* const* d_in, const int* in_sizes, int n_in,
                              void* d_out, int out_size, void* d_ws, size_t ws_size,
                              hipStream_t stream) {
  const float* x      = (const float*)d_in[0];
  const float* W_attn = (const float*)d_in[1];
  const float* b_attn = (const float*)d_in[2];
  const float* W_proj = (const float*)d_in[3];
  const float* b_proj = (const float*)d_in[4];
  float* out = (float*)d_out;

  char* ws = (char*)d_ws;
  size_t off = 0;
  auto alloc = [&](size_t bytes) -> void* {
    void* p = ws + off;
    off += (bytes + 255) & ~(size_t)255;
    return p;
  };
  const size_t qkv_bytes = (size_t)B_ * H_ * T_ * HDP_ * 2;      // 20.97 MB each
  bf16_t* xb  = (bf16_t*)alloc((size_t)M_ * KP_ * 2);            // also reused as y
  bf16_t* waT = (bf16_t*)alloc((size_t)N1_ * KP_ * 2);
  bf16_t* wpT = (bf16_t*)alloc((size_t)C_ * KP_ * 2);
  bf16_t* qb  = (bf16_t*)alloc(qkv_bytes);
  bf16_t* kb  = (bf16_t*)alloc(qkv_bytes);                       // contiguous with qb
  bf16_t* vb  = (bf16_t*)alloc(qkv_bytes);                       // contiguous with kb

  // Zero the (contiguous) q/k/v buffers so head-dim pads stay zero.
  {
    int n128 = (int)(3 * qkv_bytes / 16);
    zero_u128_k<<<(n128 + 255) / 256, 256, 0, stream>>>((u32x4*)qb, n128);
  }
  conv_x_k<<<(M_ * KP_ + 255) / 256, 256, 0, stream>>>(x, xb);
  conv_wT_k<<<(N1_ * KP_ + 255) / 256, 256, 0, stream>>>(W_attn, waT, N1_, N1_ * KP_);
  conv_wT_k<<<(C_ * KP_ + 255) / 256, 256, 0, stream>>>(W_proj, wpT, C_, C_ * KP_);

  gemm_qkv_k<<<dim3(M_ / 64, (N1_ + 255) / 256), 256, 0, stream>>>(
      xb, waT, b_attn, qb, kb, vb);

  // y aliases xb: attention overwrites cols 0..1079; pad cols stay zero.
  attn_k<<<(B_ * H_ * (T_ / 16)) / 8, 256, 0, stream>>>(qb, kb, vb, xb);

  gemm_proj_k<<<dim3(M_ / 64, (C_ + 255) / 256), 256, 0, stream>>>(
      xb, wpT, b_proj, out);
}
